// TextRNN_14482629722469
// MI455X (gfx1250) — matrix operand order repaired and checked by reading, not compiled, not run on hardware
//
#include <hip/hip_runtime.h>
#include <cstdint>
#include <cstddef>

// ---------------------------------------------------------------------------
// CDNA5 (gfx1250) TextRNN: bf16 WMMA GEMMs + LDS-resident recurrence.
// Shapes: B=64, T=128, C=10000, H=512.
// v3: fp32->bf16 via v_perm_b32 packing (no bf16-cvt builtin on this
//     toolchain): 3 VALU per pair instead of ~9. Round-to-nearest(away).
// ---------------------------------------------------------------------------

typedef __bf16 bf16_t;
typedef bf16_t v16bf __attribute__((ext_vector_type(16)));
typedef float  v8f   __attribute__((ext_vector_type(8)));

static constexpr int Bsz  = 64;
static constexpr int Tsz  = 128;
static constexpr int Csz  = 10000;
static constexpr int Cpad = 10016;   // Csz rounded up to 32 (K-tile), zero-padded
static constexpr int Hsz  = 512;

// fp32 -> bf16 bits, round-to-nearest (ties away): add 0x8000, keep high half.
__device__ __forceinline__ uint16_t f2bfu(float f) {
    return (uint16_t)((__float_as_uint(f) + 0x8000u) >> 16);
}

// two fp32 -> packed bf16 pair: 2x v_add + 1x v_perm_b32.
// perm concat = {src0, src1}; bytes 0-3 = src1, 4-7 = src0.
// result: [1:0] = lo bytes[3:2], [3:2] = hi bytes[3:2]  -> sel 0x07060302.
__device__ __forceinline__ uint32_t cvt2(float lo, float hi) {
    const uint32_t ulo = __float_as_uint(lo) + 0x8000u;
    const uint32_t uhi = __float_as_uint(hi) + 0x8000u;
    return __builtin_amdgcn_perm(uhi, ulo, 0x07060302u);
}

union Frag {                 // reinterpret raw bf16 memory as a WMMA fragment
    v16bf    v;
    uint32_t d[8];
    uint4    q[2];
};

#define WMMA_BF16(acc, a, bfr) \
    __builtin_amdgcn_wmma_f32_16x16x32_bf16(false, (a), false, (bfr), (short)0, (acc), false, false)

// ---------------------------------------------------------------------------
// One-shot weight conversions (bf16 staging).
// ---------------------------------------------------------------------------

// W_ih (H x C f32) -> W_ihb (H x Cpad bf16), pad columns zeroed.
__global__ __launch_bounds__(256) void k_cvt_wih(const float* __restrict__ W_ih,
                                                 uint16_t* __restrict__ W_ihb)
{
    const size_t idx = (size_t)blockIdx.x * 256 + threadIdx.x;
    if (idx >= (size_t)Hsz * Cpad) return;
    const int h = (int)(idx / Cpad);
    const int c = (int)(idx % Cpad);
    W_ihb[idx] = (c < Csz) ? f2bfu(W_ih[(size_t)h * Csz + c]) : (uint16_t)0;
}

// W_hh (H x H f32) -> W_hhb (bf16), same layout.
__global__ __launch_bounds__(256) void k_cvt_whh(const float* __restrict__ W_hh,
                                                 uint16_t* __restrict__ W_hhb)
{
    const int idx = blockIdx.x * 256 + threadIdx.x;
    if (idx < Hsz * Hsz) W_hhb[idx] = f2bfu(W_hh[idx]);
}

// W (1024 x 10000 f32) -> Wt (10000 x 1024 bf16), tiled transpose
// (coalesced reads along c, coalesced writes along j).
__global__ __launch_bounds__(256) void k_transpose_w(const float* __restrict__ W,
                                                     uint16_t* __restrict__ Wt)
{
    __shared__ uint16_t tile[32][33];
    const int c0 = blockIdx.x * 32;            // C dimension base
    const int j0 = blockIdx.y * 32;            // 2H dimension base (1024 % 32 == 0)
    const int tx = threadIdx.x & 31;
    const int ty = threadIdx.x >> 5;           // 0..7
#pragma unroll
    for (int r = 0; r < 32; r += 8) {
        const int j = j0 + ty + r;
        const int c = c0 + tx;
        tile[ty + r][tx] = (c < Csz) ? f2bfu(W[(size_t)j * Csz + c]) : (uint16_t)0;
    }
    __syncthreads();
#pragma unroll
    for (int r = 0; r < 32; r += 8) {
        const int c = c0 + ty + r;
        const int j = j0 + tx;
        if (c < Csz) Wt[(size_t)c * (2 * Hsz) + j] = tile[tx][ty + r];
    }
}

// ---------------------------------------------------------------------------
// Kernel 1: pre[t*B+b][h] = X[b][t][:] . W_ih[h][:] + b_ih[h] + b_hh[h]
// Block = 8 waves; wave w owns N columns [w*64, w*64+64) => the block covers
// ALL of N for its 16 M rows, so X is streamed from HBM exactly once
// (intra-block re-reads are WGP$/L2 hits). grid = M/16 = 512 blocks.
// HBM floor: 328 MB of X @ 23.3 TB/s ~= 14 us; conversion VALU co-executes
// with the WMMA (XDL) pipe, so the loop stays load/WMMA bound.
// ---------------------------------------------------------------------------
__global__ __launch_bounds__(256) void k_gemm_pre(const float* __restrict__ X,
                                                  const uint16_t* __restrict__ W_ihb,
                                                  const float* __restrict__ b_ih,
                                                  const float* __restrict__ b_hh,
                                                  float* __restrict__ pre)
{
    const int lane = threadIdx.x & 31;
    const int wave = threadIdx.x >> 5;     // 0..7 -> N group
    const int hf   = lane >> 4;
    const int l16  = lane & 15;
    const int mt   = blockIdx.x;
    const int nt0  = wave * 4;

    // A-matrix row: m = t*B + b  ->  X[b][t][:]
    const int m = mt * 16 + l16;
    const int t = m >> 6;
    const int b = m & 63;
    const float* __restrict__ xrow = X + ((size_t)b * Tsz + t) * Csz;

    const int abase = hf * 8;    // A K-offsets: {0..7,16..23} / {8..15,24..31}
    const int bbase = hf * 8;    // B K-offsets: {0..7,16..23} / {8..15,24..31}

    v8f acc[4] = {};

    int kk = 0;
    for (; kk + 32 <= Csz; kk += 32) {
        __builtin_prefetch(xrow + kk + 512, 0, 1);   // global_prefetch_b8
        const float* xp = xrow + kk + abase;
        const float4 x0 = *(const float4*)(xp);
        const float4 x1 = *(const float4*)(xp + 4);
        const float4 x2 = *(const float4*)(xp + 16);
        const float4 x3 = *(const float4*)(xp + 20);
        Frag a;
        a.d[0] = cvt2(x0.x, x0.y);  a.d[1] = cvt2(x0.z, x0.w);
        a.d[2] = cvt2(x1.x, x1.y);  a.d[3] = cvt2(x1.z, x1.w);
        a.d[4] = cvt2(x2.x, x2.y);  a.d[5] = cvt2(x2.z, x2.w);
        a.d[6] = cvt2(x3.x, x3.y);  a.d[7] = cvt2(x3.z, x3.w);
#pragma unroll
        for (int j = 0; j < 4; ++j) {
            const int n = (nt0 + j) * 16 + l16;
            const uint4* wp = (const uint4*)(W_ihb + (size_t)n * Cpad + kk + bbase);
            Frag bf_;
            bf_.q[0] = wp[0];
            bf_.q[1] = wp[2];
            acc[j] = WMMA_BF16(acc[j], a.v, bf_.v);
        }
    }
    // K tail (Csz % 32 == 16): both half-waves have their first 8 A elements
    // in-range and last 8 out-of-range -> zero upper dwords, no per-elt guards.
    // B side reads the zero-padded columns of W_ihb (no guards either).
    {
        const float* xp = xrow + kk + abase;
        const float4 x0 = *(const float4*)(xp);
        const float4 x1 = *(const float4*)(xp + 4);
        Frag a;
        a.d[0] = cvt2(x0.x, x0.y);  a.d[1] = cvt2(x0.z, x0.w);
        a.d[2] = cvt2(x1.x, x1.y);  a.d[3] = cvt2(x1.z, x1.w);
        a.d[4] = 0; a.d[5] = 0; a.d[6] = 0; a.d[7] = 0;
#pragma unroll
        for (int j = 0; j < 4; ++j) {
            const int n = (nt0 + j) * 16 + l16;
            const uint4* wp = (const uint4*)(W_ihb + (size_t)n * Cpad + kk + bbase);
            Frag bf_;
            bf_.q[0] = wp[0];
            bf_.q[1] = wp[2];
            acc[j] = WMMA_BF16(acc[j], a.v, bf_.v);
        }
    }
    // D layout: lane L, VGPR v -> (M = v + 8*(L>=16), N = L%16)
#pragma unroll
    for (int j = 0; j < 4; ++j) {
        const int N = (nt0 + j) * 16 + l16;
        const float bias = b_ih[N] + b_hh[N];
#pragma unroll
        for (int v = 0; v < 8; ++v) {
            const int M = mt * 16 + hf * 8 + v;
            pre[(size_t)M * Hsz + N] = acc[j][v] + bias;
        }
    }
}

// ---------------------------------------------------------------------------
// Kernel 2: the sequential recurrence. One 1024-thread workgroup (32 waves)
// on one WGP; h kept in LDS as bf16 with padded row stride (520 elts) so the
// 16 A-fragment rows hit disjoint bank groups. Wave w owns column-tile w
// (16 cols) x all 4 M tiles: B fragment loaded once per K step, reused 4x.
// ---------------------------------------------------------------------------
__global__ __launch_bounds__(1024) void k_rnn_scan(const float* __restrict__ pre,
                                                   const float* __restrict__ hidden,
                                                   const uint16_t* __restrict__ W_hhb,
                                                   float* __restrict__ outputs)
{
    constexpr int LP = Hsz + 8;             // padded LDS row stride (elements)
    __shared__ uint16_t h_lds[Bsz * LP];    // 66,560 B (of 320 KB)

    const int tid  = threadIdx.x;
    const int lane = tid & 31;
    const int wave = tid >> 5;              // 0..31 = N column tile
    const int hf   = lane >> 4;
    const int l16  = lane & 15;

    for (int i = tid; i < Bsz * Hsz; i += 1024)
        h_lds[(i >> 9) * LP + (i & (Hsz - 1))] = f2bfu(hidden[i]);
    __syncthreads();

    const int ncol = wave * 16 + l16;       // j index into W_hh rows (output col)
    const uint16_t* __restrict__ wrow = W_hhb + (size_t)ncol * Hsz;
    const int abase = hf * 8;
    const int bbase = hf * 8;

    for (int t = 0; t < Tsz; ++t) {
        v8f acc[4] = {};
#pragma unroll 4
        for (int kk = 0; kk < Hsz; kk += 32) {
            const uint4* wp = (const uint4*)(wrow + kk + bbase);
            Frag bf_;
            bf_.q[0] = wp[0];
            bf_.q[1] = wp[2];
#pragma unroll
            for (int j = 0; j < 4; ++j) {
                const uint16_t* hp = h_lds + (j * 16 + l16) * LP + kk + abase;
                Frag a;
                a.q[0] = *(const uint4*)(hp);
                a.q[1] = *(const uint4*)(hp + 16);
                acc[j] = WMMA_BF16(acc[j], a.v, bf_.v);
            }
        }
        __syncthreads();   // all LDS reads of h done before overwrite

        const float* __restrict__ pre_t = pre + (size_t)t * Bsz * Hsz;
        float* __restrict__ out_t = outputs + (size_t)t * Bsz * Hsz;
#pragma unroll
        for (int j = 0; j < 4; ++j) {
#pragma unroll
            for (int v = 0; v < 8; ++v) {
                const int M = j * 16 + hf * 8 + v;          // batch row
                const float val = tanhf(acc[j][v] + pre_t[M * Hsz + ncol]);
                out_t[M * Hsz + ncol] = val;
                h_lds[M * LP + ncol] = f2bfu(val);
            }
        }
        __syncthreads();   // h_new visible to all waves
    }
}

// ---------------------------------------------------------------------------
// Kernel 3: attention over time + build the bf16 concat matrix.
// ---------------------------------------------------------------------------
__global__ __launch_bounds__(128) void k_attn(const float* __restrict__ outputs,
                                              uint16_t* __restrict__ catb)
{
    __shared__ float sc[128];
    __shared__ float red[128];

    const int b   = blockIdx.x;
    const int tid = threadIdx.x;
    const float* __restrict__ last = outputs + (size_t)(Tsz - 1) * Bsz * Hsz + (size_t)b * Hsz;

    if (tid < Tsz - 1) {
        const float* __restrict__ prow = outputs + (size_t)tid * Bsz * Hsz + (size_t)b * Hsz;
        float s = 0.f;
        for (int h = 0; h < Hsz; ++h) s += prow[h] * last[h];
        sc[tid] = s;
    }
    __syncthreads();

    red[tid] = (tid < Tsz - 1) ? sc[tid] : -3.0e38f;
    __syncthreads();
    for (int s = 64; s; s >>= 1) {
        if (tid < s) red[tid] = fmaxf(red[tid], red[tid + s]);
        __syncthreads();
    }
    const float mx = red[0];
    __syncthreads();
    const float e = (tid < Tsz - 1) ? __expf(sc[tid] - mx) : 0.f;
    red[tid] = e;
    __syncthreads();
    for (int s = 64; s; s >>= 1) {
        if (tid < s) red[tid] += red[tid + s];
        __syncthreads();
    }
    const float inv = 1.f / red[0];
    __syncthreads();
    if (tid < Tsz - 1) sc[tid] = e * inv;
    __syncthreads();

    for (int h = tid; h < Hsz; h += 128) {
        float s = 0.f;
        for (int t = 0; t < Tsz - 1; ++t)
            s += sc[t] * outputs[(size_t)t * Bsz * Hsz + (size_t)b * Hsz + h];
        catb[(size_t)b * (2 * Hsz) + h]       = f2bfu(s);
        catb[(size_t)b * (2 * Hsz) + Hsz + h] = f2bfu(last[h]);
    }
}

// ---------------------------------------------------------------------------
// Kernel 5: out[b][c] = cat[b][:] . W[:][c] + bias[c].  One wave per block
// computes 64(M) x 16(N): B fragment loaded once per K step, reused 4x.
// grid = C/16 = 625.  All operands already bf16 -> pure load+wmma loop.
// ---------------------------------------------------------------------------
__global__ __launch_bounds__(32) void k_gemm_out(const uint16_t* __restrict__ catb,
                                                 const uint16_t* __restrict__ Wt,
                                                 const float* __restrict__ bias,
                                                 float* __restrict__ out)
{
    const int lane = threadIdx.x & 31;
    const int hf   = lane >> 4;
    const int l16  = lane & 15;
    const int n    = blockIdx.x * 16 + l16;
    const uint16_t* __restrict__ wtrow = Wt + (size_t)n * (2 * Hsz);
    const int abase = hf * 8;
    const int bbase = hf * 8;

    v8f acc[4] = {};
#pragma unroll 4
    for (int kk = 0; kk < 2 * Hsz; kk += 32) {
        const uint4* wp = (const uint4*)(wtrow + kk + bbase);
        Frag bf_;
        bf_.q[0] = wp[0];
        bf_.q[1] = wp[2];
#pragma unroll
        for (int j = 0; j < 4; ++j) {
            const uint16_t* ap = catb + (size_t)(j * 16 + l16) * (2 * Hsz) + kk + abase;
            Frag a;
            a.q[0] = *(const uint4*)(ap);
            a.q[1] = *(const uint4*)(ap + 16);
            acc[j] = WMMA_BF16(acc[j], a.v, bf_.v);
        }
    }
#pragma unroll
    for (int j = 0; j < 4; ++j) {
#pragma unroll
        for (int v = 0; v < 8; ++v) {
            const int M = j * 16 + hf * 8 + v;   // batch row
            out[(size_t)M * Csz + n] = acc[j][v] + bias[n];
        }
    }
}

// ---------------------------------------------------------------------------
// Launch. Workspace layout (bytes):
//   [0,            16777216)  pre      (8192 x 512 f32)
//   [16777216,     33554432)  outputs  (128 x 64 x 512 f32)
//   [33554432,     33685504)  cat      (64 x 1024 bf16)
//   [33685504,     54165504)  Wt       (10000 x 1024 bf16)
//   [54165504,     64421888)  W_ihb    (512 x 10016 bf16, K-padded)
//   [64421888,     64946176)  W_hhb    (512 x 512 bf16)
// ---------------------------------------------------------------------------
extern "C" void kernel_launch(void* const* d_in, const int* in_sizes, int n_in,
                              void* d_out, int out_size, void* d_ws, size_t ws_size,
                              hipStream_t stream)
{
    const float* X      = (const float*)d_in[0];
    const float* hidden = (const float*)d_in[1];
    const float* W_ih   = (const float*)d_in[2];
    const float* W_hh   = (const float*)d_in[3];
    const float* b_ih   = (const float*)d_in[4];
    const float* b_hh   = (const float*)d_in[5];
    const float* W      = (const float*)d_in[6];
    const float* bias   = (const float*)d_in[7];
    float* out = (float*)d_out;

    char* ws = (char*)d_ws;
    float*    pre     = (float*)(ws + 0);
    float*    outputs = (float*)(ws + 16777216);
    uint16_t* catb    = (uint16_t*)(ws + 33554432);
    uint16_t* Wt      = (uint16_t*)(ws + 33685504);
    uint16_t* W_ihb   = (uint16_t*)(ws + 54165504);
    uint16_t* W_hhb   = (uint16_t*)(ws + 64421888);

    // One-shot bf16 staging of all weights.
    k_cvt_wih<<<(Hsz * Cpad + 255) / 256, 256, 0, stream>>>(W_ih, W_ihb);
    k_cvt_whh<<<(Hsz * Hsz + 255) / 256, 256, 0, stream>>>(W_hh, W_hhb);
    k_transpose_w<<<dim3((Csz + 31) / 32, (2 * Hsz) / 32), 256, 0, stream>>>(W, Wt);

    // Big streaming GEMM: 512 blocks x 8 waves (full N per block).
    k_gemm_pre<<<Tsz * Bsz / 16, 256, 0, stream>>>(X, W_ihb, b_ih, b_hh, pre);

    // Sequential recurrence: one WGP, 32 waves, h in LDS.
    k_rnn_scan<<<1, 1024, 0, stream>>>(pre, hidden, W_hhb, outputs);

    // Attention + concat (bf16).
    k_attn<<<Bsz, 128, 0, stream>>>(outputs, catb);

    // Final projection GEMM: 625 single-wave blocks.
    k_gemm_out<<<Csz / 16, 32, 0, stream>>>(catb, Wt, bias, out);
}
